// CrossModelAttNew_824633721620
// MI455X (gfx1250) — matrix operand order repaired and checked by reading, not compiled, not run on hardware
//
#include <hip/hip_runtime.h>
#include <hip/hip_bf16.h>
#include <stdint.h>

typedef __attribute__((ext_vector_type(16))) __bf16 v16bf;
typedef __attribute__((ext_vector_type(8)))  float  v8f;

#define B_  32
#define C_  768
#define S_  1024
#define L_  77
#define D_  512

#define TM 128
#define TN 64
#define TK 32

#define APAD 4   // fp32 row pad (16B): keeps 16B alignment, spreads banks (36*m mod 64)
#define BPAD 8   // bf16 row pad (16B): keeps 16B alignment, spreads banks (20*n mod 64)

// ---- CDNA5 async copy: global -> LDS, 16 bytes per lane, tracked by ASYNCcnt
__device__ __forceinline__ void async_b128(uint32_t lds, uint64_t gaddr) {
    asm volatile("global_load_async_to_lds_b128 %0, %1, off"
                 :: "v"(lds), "v"(gaddr) : "memory");
}
__device__ __forceinline__ void wait_async0() {
    asm volatile("s_wait_asynccnt 0x0" ::: "memory");
}
__device__ __forceinline__ uint32_t lds_addr(const void* p) {
    // flat shared address truncates to the LDS byte offset (ISA 10.2 aperture table)
    return (uint32_t)(uintptr_t)p;
}

// ---------------- Kernel 1: text projection  tproj[b][c][l] = sum_d text[b,l,d]*W[d,c]
__global__ __launch_bounds__(256) void proj_kernel(const float* __restrict__ text,
                                                   const float* __restrict__ Wtxt,
                                                   float* __restrict__ tproj) {
    int c = blockIdx.x * blockDim.x + threadIdx.x;   // coalesced over W columns
    int l = blockIdx.y;
    int b = blockIdx.z;
    if (c >= C_) return;
    const float* tv = text + ((size_t)b * L_ + l) * D_;
    float acc = 0.0f;
    #pragma unroll 8
    for (int d = 0; d < D_; ++d)
        acc = fmaf(tv[d], Wtxt[(size_t)d * C_ + c], acc);
    tproj[((size_t)b * C_ + c) * L_ + l] = acc;
}

// ---------------- Kernel 2: linear interpolation L=77 -> S=1024, emit bf16 t[b][c][s]
__global__ __launch_bounds__(256) void interp_kernel(const float* __restrict__ tproj,
                                                     __bf16* __restrict__ tbf) {
    int s = blockIdx.x * blockDim.x + threadIdx.x;
    int c = blockIdx.y;
    int b = blockIdx.z;
    float src = (s + 0.5f) * ((float)L_ / (float)S_) - 0.5f;
    src = fminf(fmaxf(src, 0.0f), (float)(L_ - 1));
    int i0 = (int)src;
    int i1 = i0 + 1; if (i1 > L_ - 1) i1 = L_ - 1;
    float w = src - (float)i0;
    const float* row = tproj + ((size_t)b * C_ + c) * L_;
    float v = row[i0] * (1.0f - w) + row[i1] * w;
    tbf[((size_t)b * C_ + c) * S_ + s] = (__bf16)v;
}

// ---------------- Kernel 3: logits[b][i][j] = (1/32) * sum_s q[b,i,s]*t[b,j,s]
// 256 thr = 8 waves; tile 128(i) x 64(j); K-step 32; double-buffered async LDS staging.
__global__ __launch_bounds__(256) void attn_gemm_kernel(const float* __restrict__ q,
                                                        const __bf16* __restrict__ tbf,
                                                        float* __restrict__ logits) {
    __shared__ __align__(16) float  As[2][TM][TK + APAD];   // q tile fp32
    __shared__ __align__(16) __bf16 Bs[2][TN][TK + BPAD];   // t tile bf16

    int b    = blockIdx.z;
    int row0 = blockIdx.x * TM;
    int col0 = blockIdx.y * TN;
    int tid  = threadIdx.x;
    int lane = tid & 31;
    int wave = tid >> 5;

    const float*  qb = q   + (size_t)b * C_ * S_;
    const __bf16* tb = tbf + (size_t)b * C_ * S_;

    auto issue = [&](int buf, int kk) {
        // A tile: 128x32 fp32 = 1024 16B-chunks -> 4 per thread
        #pragma unroll
        for (int i = 0; i < 4; ++i) {
            int id = tid + 256 * i;
            int r = id >> 3, c4 = (id & 7) * 4;
            async_b128(lds_addr(&As[buf][r][c4]),
                       (uint64_t)(uintptr_t)&qb[(size_t)(row0 + r) * S_ + kk + c4]);
        }
        // B tile: 64x32 bf16 = 256 16B-chunks -> 1 per thread
        int r = tid >> 2, c8 = (tid & 3) * 8;
        async_b128(lds_addr(&Bs[buf][r][c8]),
                   (uint64_t)(uintptr_t)&tb[(size_t)(col0 + r) * S_ + kk + c8]);
    };

    v8f acc[4] = {};
    int mloc  = lane & 15;
    int khalf = lane >> 4;

    issue(0, 0);
    wait_async0();
    __syncthreads();

    int p = 0;
    for (int kk = 0; kk < S_; kk += TK) {
        int np = p ^ 1;
        if (kk + TK < S_) issue(np, kk + TK);      // overlap copy with compute

        // A fragment: one contiguous 64B run per lane (fp32 -> bf16 in-register)
        const float* ar = &As[p][wave * 16 + mloc][khalf * 16];
        v16bf a;
        #pragma unroll
        for (int j = 0; j < 8; ++j) {
            a[j]     = (__bf16)ar[j];
            a[8 + j] = (__bf16)ar[8 + j];
        }
        #pragma unroll
        for (int t = 0; t < 4; ++t) {
            // B fragment: one contiguous 32B run per lane
            const __bf16* br = &Bs[p][t * 16 + mloc][khalf * 16];
            v16bf bb;
            #pragma unroll
            for (int j = 0; j < 16; ++j) bb[j] = br[j];
            acc[t] = __builtin_amdgcn_wmma_f32_16x16x32_bf16(
                false, a, false, bb, (short)0, acc[t], false, false);
        }
        wait_async0();       // next tile landed (this wave's chunks)
        __syncthreads();     // all waves done with buf p, see buf np
        p = np;
    }

    const float scale = 0.03125f;    // S^-0.5
    int mhi = khalf * 8;
    #pragma unroll
    for (int t = 0; t < 4; ++t)
        #pragma unroll
        for (int v = 0; v < 8; ++v) {
            int r  = row0 + wave * 16 + v + mhi;
            int cc = col0 + t * 16 + mloc;
            logits[((size_t)b * C_ + r) * C_ + cc] = acc[t][v] * scale;
        }
}

// ---------------- Kernel 4: row softmax over 768, emit bf16 attn
__global__ __launch_bounds__(256) void softmax_kernel(const float* __restrict__ logits,
                                                      __bf16* __restrict__ attn) {
    int row = blockIdx.x;
    const float* src = logits + (size_t)row * C_;
    __bf16* dst = attn + (size_t)row * C_;
    __shared__ float red[256];
    int tid = threadIdx.x;

    float v0 = src[tid], v1 = src[tid + 256], v2 = src[tid + 512];
    float mx = fmaxf(v0, fmaxf(v1, v2));
    red[tid] = mx; __syncthreads();
    for (int off = 128; off; off >>= 1) {
        if (tid < off) red[tid] = fmaxf(red[tid], red[tid + off]);
        __syncthreads();
    }
    mx = red[0]; __syncthreads();

    v0 = __expf(v0 - mx); v1 = __expf(v1 - mx); v2 = __expf(v2 - mx);
    red[tid] = v0 + v1 + v2; __syncthreads();
    for (int off = 128; off; off >>= 1) {
        if (tid < off) red[tid] += red[tid + off];
        __syncthreads();
    }
    float inv = 1.0f / red[0];
    dst[tid]       = (__bf16)(v0 * inv);
    dst[tid + 256] = (__bf16)(v1 * inv);
    dst[tid + 512] = (__bf16)(v2 * inv);
}

// ---------------- Kernel 5: out[b][i][s] = q[b,i,s] + gamma * sum_j attn[i,j]*t[j,s]
__global__ __launch_bounds__(256) void out_gemm_kernel(const __bf16* __restrict__ attn,
                                                       const __bf16* __restrict__ tbf,
                                                       const float* __restrict__ q,
                                                       const float* __restrict__ gamma,
                                                       float* __restrict__ out) {
    __shared__ __align__(16) __bf16 As [2][TM][TK + BPAD];  // attn tile (row-major, async)
    __shared__ __align__(16) __bf16 Bsx[2][TN][TK + BPAD];  // t tile TRANSPOSED: [s-local][j-local]

    int b    = blockIdx.z;
    int row0 = blockIdx.x * TM;   // i
    int col0 = blockIdx.y * TN;   // s
    int tid  = threadIdx.x;
    int lane = tid & 31;
    int wave = tid >> 5;

    const __bf16* ab = attn + (size_t)b * C_ * C_;
    const __bf16* tb = tbf  + (size_t)b * C_ * S_;

    auto issueA = [&](int buf, int kk) {
        // A tile: 128x32 bf16 = 512 16B-chunks -> 2 per thread
        #pragma unroll
        for (int i = 0; i < 2; ++i) {
            int id = tid + 256 * i;
            int r = id >> 2, c8 = (id & 3) * 8;
            async_b128(lds_addr(&As[buf][r][c8]),
                       (uint64_t)(uintptr_t)&ab[(size_t)(row0 + r) * C_ + kk + c8]);
        }
    };
    auto stageB = [&](int buf, int kk) {
        // read coalesced over s, store transposed so frag reads are contiguous
        #pragma unroll
        for (int i = 0; i < 8; ++i) {
            int id = tid + 256 * i;
            int r = id >> 6, cc = id & 63;      // r = j-local (0..31), cc = s-local (0..63)
            Bsx[buf][cc][r] = tb[(size_t)(kk + r) * S_ + col0 + cc];
        }
    };

    v8f acc[4] = {};
    int mloc  = lane & 15;
    int khalf = lane >> 4;

    issueA(0, 0);
    stageB(0, 0);
    wait_async0();
    __syncthreads();

    int p = 0;
    for (int kk = 0; kk < C_; kk += TK) {
        int np = p ^ 1;
        if (kk + TK < C_) { issueA(np, kk + TK); stageB(np, kk + TK); }

        const __bf16* ar = &As[p][wave * 16 + mloc][khalf * 16];
        v16bf a;
        #pragma unroll
        for (int j = 0; j < 8; ++j) {
            a[j]     = ar[j];
            a[8 + j] = ar[8 + j];
        }
        #pragma unroll
        for (int t = 0; t < 4; ++t) {
            const __bf16* br = &Bsx[p][t * 16 + mloc][khalf * 16];
            v16bf bb;
            #pragma unroll
            for (int j = 0; j < 16; ++j) bb[j] = br[j];
            acc[t] = __builtin_amdgcn_wmma_f32_16x16x32_bf16(
                false, a, false, bb, (short)0, acc[t], false, false);
        }
        wait_async0();
        __syncthreads();
        p = np;
    }

    float g = gamma[0];
    const float* qb = q   + (size_t)b * C_ * S_;
    float*       ob = out + (size_t)b * C_ * S_;
    int mhi = khalf * 8;
    #pragma unroll
    for (int t = 0; t < 4; ++t)
        #pragma unroll
        for (int v = 0; v < 8; ++v) {
            int r  = row0 + wave * 16 + v + mhi;
            int cc = col0 + t * 16 + mloc;
            size_t o = (size_t)r * S_ + cc;
            ob[o] = qb[o] + g * acc[t][v];
        }
}

// ---------------- Host-side launcher
extern "C" void kernel_launch(void* const* d_in, const int* in_sizes, int n_in,
                              void* d_out, int out_size, void* d_ws, size_t ws_size,
                              hipStream_t stream) {
    const float* img   = (const float*)d_in[0];   // [B,C,H,W] == [B,C,S]
    const float* text  = (const float*)d_in[1];   // [B,L,D]
    const float* Wtxt  = (const float*)d_in[2];   // [D,C]
    const float* gamma = (const float*)d_in[3];   // [1]
    float* out = (float*)d_out;

    auto align256 = [](size_t x) { return (x + 255) & ~(size_t)255; };
    char* ws = (char*)d_ws;
    size_t off = 0;
    float*  tproj  = (float*)(ws + off);  off = align256(off + (size_t)B_ * C_ * L_ * sizeof(float));
    __bf16* tbf    = (__bf16*)(ws + off); off = align256(off + (size_t)B_ * C_ * S_ * sizeof(__bf16));
    float*  logits = (float*)(ws + off);  off = align256(off + (size_t)B_ * C_ * C_ * sizeof(float));
    __bf16* attnbf = (__bf16*)(ws + off); off = align256(off + (size_t)B_ * C_ * C_ * sizeof(__bf16));
    (void)ws_size; (void)in_sizes; (void)n_in; (void)out_size;

    proj_kernel  <<<dim3(C_ / 256, L_, B_), 256, 0, stream>>>(text, Wtxt, tproj);
    interp_kernel<<<dim3(S_ / 256, C_, B_), 256, 0, stream>>>(tproj, tbf);
    attn_gemm_kernel<<<dim3(C_ / TM, C_ / TN, B_), 256, 0, stream>>>(img, tbf, logits);
    softmax_kernel  <<<dim3(B_ * C_), 256, 0, stream>>>(logits, attnbf);
    out_gemm_kernel <<<dim3(C_ / TM, S_ / TN, B_), 256, 0, stream>>>(attnbf, tbf, img, gamma, out);
}
